// WaveletKANLayer_51067161149473
// MI455X (gfx1250) — matrix operand order repaired and checked by reading, not compiled, not run on hardware
//
#include <hip/hip_runtime.h>

// ---------------------------------------------------------------------------
// WaveletKAN on MI455X (gfx1250):
//   stage 1: pack bases -> BT[256][512] bf16 (transposed, zero-padded K)
//   stage 2: per-row 4-level db4 dwt in LDS, emit packed A[16384][512] bf16
//   stage 3: GEMM 16384x512x256 via v_wmma_f32_16x16x32_bf16 (f32 accum),
//            double-buffered fragments so loads pipeline ahead of the WMMAs
// HBM traffic ~112 MB => ~4.8us floor @ 23.3 TB/s; 3.9 GFLOP of bf16 WMMA
// is far below the matrix-core ceiling, so stage 2/3 are bandwidth-limited.
// ---------------------------------------------------------------------------

typedef __attribute__((ext_vector_type(16))) __bf16 v16bf;
typedef __attribute__((ext_vector_type(8)))  float  v8f;

#define BATCH     16384
#define KDIM      512      // 466 real + zero pad
#define NDIM      256

// dwt correlation kernels: y[i] = sum_t e[2i+t]*kern[t], kern = filt[::-1]
__device__ __constant__ float KLO[8] = {
     0.2303778133088965f,  0.7148465705529157f,  0.6308807679298589f,
    -0.027983769416859854f,-0.18703481171909309f, 0.030841381835560764f,
     0.0328830116668852f, -0.010597401785069032f };
__device__ __constant__ float KHI[8] = {
    -0.010597401785069032f,-0.0328830116668852f,  0.030841381835560764f,
     0.18703481171909309f, -0.027983769416859854f,-0.6308807679298589f,
     0.7148465705529157f,  -0.2303778133088965f };

static __device__ __forceinline__ unsigned short f2bf(float f) {
    unsigned int u = __float_as_uint(f);
    u = (u + 0x7FFFu + ((u >> 16) & 1u)) >> 16;   // round-to-nearest-even
    return (unsigned short)u;
}

// symmetric extension e[] of a length-N signal, e index j in [0, N+12]
static __device__ __forceinline__ float eread(const float* s, int N, int j) {
    int idx = (j < 6) ? (5 - j) : ((j < N + 6) ? (j - 6) : (2 * N + 5 - j));
    return s[idx];
}

// one dwt level: in[N] -> cA[outLen] (LDS), cD[:nD] -> global packed row (bf16)
static __device__ __forceinline__ void dwt_level(const float* in, int N,
                                                 float* outA, int outLen,
                                                 unsigned short* outD, int nD,
                                                 int tid, int nthr) {
    for (int i = tid; i < outLen; i += nthr) {
        float a = 0.f, d = 0.f;
#pragma unroll
        for (int t = 0; t < 8; ++t) {
            float v = eread(in, N, 2 * i + t);
            a = fmaf(v, KLO[t], a);
            d = fmaf(v, KHI[t], d);
        }
        outA[i] = a;
        if (i < nD) outD[i] = f2bf(d);
    }
}

// ---- stage 2: fused wavelet cascade + bf16 pack, one block per batch row ----
__global__ __launch_bounds__(256) void wavelet_pack_kernel(
        const float* __restrict__ x, unsigned short* __restrict__ Ap) {
    __shared__ float s0[1024];
    __shared__ float s1[520];
    const int row = blockIdx.x;
    const int tid = threadIdx.x;
    const float* xr = x + (size_t)row * 1024;
    unsigned short* arow = Ap + (size_t)row * KDIM;

    for (int i = tid; i < 1024; i += 256) s0[i] = xr[i];
    __syncthreads();
    // L1: x(1024) -> cA1(515) in s1, cD1[:64] -> cols 402..465
    dwt_level(s0, 1024, s1, 515, arow + 402, 64, tid, 256);
    __syncthreads();
    // L2: cA1(515) -> cA2(261) in s0, cD2[:128] -> cols 274..401
    dwt_level(s1, 515, s0, 261, arow + 274, 128, tid, 256);
    __syncthreads();
    // L3: cA2(261) -> cA3(134) in s1, cD3(134) -> cols 140..273
    dwt_level(s0, 261, s1, 134, arow + 140, 134, tid, 256);
    __syncthreads();
    // L4: cA3(134) -> cA4(70) in s0, cD4(70) -> cols 70..139
    dwt_level(s1, 134, s0, 70, arow + 70, 70, tid, 256);
    __syncthreads();
    // cA4 -> cols 0..69 ; zero pad cols 466..511
    if (tid < 70)  arow[tid] = f2bf(s0[tid]);
    if (tid < 46)  arow[466 + tid] = 0;
}

// ---- stage 1: pack 5 bases into BT[n][k] (transposed, bf16, zero-padded) ---
__global__ __launch_bounds__(256) void bpack_kernel(
        const float* __restrict__ b0, const float* __restrict__ b1,
        const float* __restrict__ b2, const float* __restrict__ b3,
        const float* __restrict__ b4, unsigned short* __restrict__ BT) {
    const int k = blockIdx.x;    // 0..511 packed K row
    const int n = threadIdx.x;   // 0..255 output column
    float v = 0.f;
    if      (k < 70)  v = b0[(size_t)k * NDIM + n];
    else if (k < 140) v = b1[(size_t)(k - 70) * NDIM + n];
    else if (k < 274) v = b2[(size_t)(k - 140) * NDIM + n];
    else if (k < 402) v = b3[(size_t)(k - 274) * NDIM + n];
    else if (k < 466) v = b4[(size_t)(k - 402) * NDIM + n];
    BT[(size_t)n * KDIM + k] = f2bf(v);
}

// ---- stage 3: bf16 WMMA GEMM, 4 waves/block, each wave owns a 16x64 strip --
union Frag { uint4 u[2]; v16bf v; };

static __device__ __forceinline__ void load_a(Frag& f,
        const unsigned short* __restrict__ arow, int k0, int h) {
    f.u[0] = *(const uint4*)(arow + k0 + 16 * h);       // K = k0+16h  .. +7
    f.u[1] = *(const uint4*)(arow + k0 + 16 * h + 8);   // K = k0+16h+8.. +15
}
static __device__ __forceinline__ void load_b4(Frag* f,
        const unsigned short* __restrict__ BT, int nb, int mr, int k0, int h) {
#pragma unroll
    for (int t = 0; t < 4; ++t) {
        const unsigned short* brow =
            BT + (size_t)(nb + 16 * t + mr) * KDIM + k0 + 16 * h;
        f[t].u[0] = *(const uint4*)(brow);              // K = k0+16h  .. +7
        f[t].u[1] = *(const uint4*)(brow + 8);          // K = k0+16h+8.. +15
    }
}
static __device__ __forceinline__ void wmma4(v8f* acc, const Frag& a,
                                             const Frag* b) {
#pragma unroll
    for (int t = 0; t < 4; ++t)
        acc[t] = __builtin_amdgcn_wmma_f32_16x16x32_bf16(
            false, a.v, false, b[t].v, (short)0, acc[t], false, false);
}

__global__ __launch_bounds__(128) void wkan_gemm_kernel(
        const unsigned short* __restrict__ A,   // [BATCH][KDIM] bf16
        const unsigned short* __restrict__ BT,  // [NDIM][KDIM]  bf16
        float* __restrict__ out) {              // [BATCH][NDIM] f32
    const int m0   = blockIdx.x * 16;
    const int wave = threadIdx.x >> 5;
    const int lane = threadIdx.x & 31;
    const int mr   = lane & 15;   // row within tile (A/B column select)
    const int h    = lane >> 4;   // lane half -> K sub-slice
    const int nb   = wave * 64;

    v8f acc[4] = {v8f{}, v8f{}, v8f{}, v8f{}};
    const unsigned short* arow = A + (size_t)(m0 + mr) * KDIM;

    // double-buffered software pipeline: loads run one k-step ahead of math
    Frag a0, a1, b0[4], b1[4];
    load_a(a0, arow, 0, h);
    load_b4(b0, BT, nb, mr, 0, h);

#pragma unroll
    for (int k0 = 0; k0 < KDIM; k0 += 64) {
        load_a(a1, arow, k0 + 32, h);          // prefetch step k0+32
        load_b4(b1, BT, nb, mr, k0 + 32, h);
        wmma4(acc, a0, b0);                    // consume step k0
        if (k0 + 64 < KDIM) {
            load_a(a0, arow, k0 + 64, h);      // prefetch step k0+64
            load_b4(b0, BT, nb, mr, k0 + 64, h);
        }
        wmma4(acc, a1, b1);                    // consume step k0+32
    }

#pragma unroll
    for (int t = 0; t < 4; ++t) {
        const int n = nb + 16 * t + mr;
        float* op = out + (size_t)(m0 + 8 * h) * NDIM + n;
#pragma unroll
        for (int r = 0; r < 8; ++r) op[(size_t)r * NDIM] = acc[t][r];
    }
}

extern "C" void kernel_launch(void* const* d_in, const int* in_sizes, int n_in,
                              void* d_out, int out_size, void* d_ws, size_t ws_size,
                              hipStream_t stream) {
    const float* x  = (const float*)d_in[0];
    const float* b0 = (const float*)d_in[1];
    const float* b1 = (const float*)d_in[2];
    const float* b2 = (const float*)d_in[3];
    const float* b3 = (const float*)d_in[4];
    const float* b4 = (const float*)d_in[5];
    float* out = (float*)d_out;

    unsigned short* Ap = (unsigned short*)d_ws;                       // 16 MB
    unsigned short* BT = (unsigned short*)((char*)d_ws +
                         (size_t)BATCH * KDIM * sizeof(unsigned short)); // 256 KB

    bpack_kernel<<<KDIM, NDIM, 0, stream>>>(b0, b1, b2, b3, b4, BT);
    wavelet_pack_kernel<<<BATCH, 256, 0, stream>>>(x, Ap);
    wkan_gemm_kernel<<<BATCH / 16, 128, 0, stream>>>(Ap, BT, out);
}
